// RoughScorer_72825465471242
// MI455X (gfx1250) — matrix operand order repaired and checked by reading, not compiled, not run on hardware
//
#include <hip/hip_runtime.h>
#include <hip/hip_bf16.h>

#define TOPK 50
#define CHUNK 512
#define LDS_STRIDE 516   // 516 % 64 = 4; rows 8 apart land 32 banks apart -> no conflict
#define F_MAX 1024

typedef __attribute__((ext_vector_type(16))) __bf16 bf16x16;
typedef __attribute__((ext_vector_type(8)))  __bf16 bf16x8;
typedef __attribute__((ext_vector_type(8)))  float  f32x8;

union FragU { bf16x16 v; bf16x8 h[2]; };

// ---- fragment loaders per CDNA5 ISA 7.12.2 ----------------------
// A (16x32 bf16, MxK): lane l -> row M = l&15; lanes 0-15 hold K {0..7,16..23},
// lanes 16-31 hold K {8..15,24..31}.
__device__ __forceinline__ bf16x16 load_a_frag(const __bf16* A, int ld,
                                               int row0, int kb, int lane) {
  int m = lane & 15, hi = lane >> 4;
  const __bf16* p = A + (size_t)(row0 + m) * ld + (kb + 8 * hi);
  FragU f;
  f.h[0] = *(const bf16x8*)(p);        // K = kb + 8*hi .. +7
  f.h[1] = *(const bf16x8*)(p + 16);   // K = kb + 16 + 8*hi .. +7
  return f.v;
}

// B (32x16 bf16, KxN) where B[k][n] = Mrows[n][k] (row-major source rows are
// the needed columns): lane l -> col N = l&15; lanes 0-15 hold K {0..7,16..23},
// lanes 16-31 hold K {8..15,24..31}.
__device__ __forceinline__ bf16x16 load_b_frag(const __bf16* Mrows, int ld,
                                               int col0, int kb, int lane) {
  int n = lane & 15, hi = lane >> 4;
  const __bf16* p = Mrows + (size_t)(col0 + n) * ld + (kb + 8 * hi);
  FragU f;
  f.h[0] = *(const bf16x8*)(p);
  f.h[1] = *(const bf16x8*)(p + 16);
  return f.v;
}

#define WMMA_BF16(a, b, c) \
  __builtin_amdgcn_wmma_f32_16x16x32_bf16(false, (a), false, (b), (short)0, (c), false, false)

// async copy of `n16` 16-byte chunks from contiguous global `gsrc` into LDS
// `ldst` using GLOBAL_LOAD_ASYNC_TO_LDS_B128 (ASYNCcnt), then wait + barrier.
__device__ __forceinline__ void async_slab_to_lds(const __bf16* gsrc, __bf16* ldst,
                                                  int n16, int tid, int nthreads) {
  for (int c = tid; c < n16; c += nthreads) {
    unsigned long long g = (unsigned long long)(uintptr_t)(gsrc + 8 * c);
    unsigned int       l = (unsigned int)(uintptr_t)(ldst + 8 * c);
    asm volatile("global_load_async_to_lds_b128 %0, %1, off"
                 :: "v"(l), "v"(g) : "memory");
  }
  asm volatile("s_wait_asynccnt 0x0" ::: "memory");
  __syncthreads();
}

// ---- kernel 1: fp32 -> bf16 conversion (mentions + W) ----------------------
__global__ void k_convert(const float* __restrict__ M, const float* __restrict__ W,
                          __bf16* __restrict__ Mb, __bf16* __restrict__ Wb,
                          long nM, long nW) {
  long i = (long)blockIdx.x * blockDim.x + threadIdx.x;
  if (i < nM) Mb[i] = (__bf16)M[i];
  if (i < nW) Wb[i] = (__bf16)W[i];
}

// ---- kernel 2: proj = mentions @ W^T + b  (bf16 WMMA, fp32 acc) ------------
// grid = (N/16, F/256), block = 128 (4 waves); each wave owns a 16x64 strip.
// The shared 16-row A slab is async-staged into LDS once per block.
__global__ __launch_bounds__(128)
void k_gemm1(const __bf16* __restrict__ Mb, const __bf16* __restrict__ Wb,
             const float* __restrict__ bias, __bf16* __restrict__ Pb,
             int N, int F) {
  int lane = threadIdx.x & 31;
  int wave = threadIdx.x >> 5;
  int mr = blockIdx.x * 16;
  int group = blockIdx.y * 4 + wave;   // 0 .. F/64-1
  int nb = group * 64;

  __shared__ __align__(16) unsigned char a_arena[16 * F_MAX * 2];  // 32 KB
  __bf16* Alds = (__bf16*)a_arena;
  async_slab_to_lds(Mb + (size_t)mr * F, Alds, (16 * F) / 8,
                    (int)threadIdx.x, 128);

  f32x8 acc[4] = {};
  for (int k0 = 0; k0 < F; k0 += 32) {
    bf16x16 a = load_a_frag(Alds, F, 0, k0, lane);     // ds_load_b128 x2
#pragma unroll
    for (int t = 0; t < 4; ++t) {
      bf16x16 bf = load_b_frag(Wb, F, nb + 16 * t, k0, lane);  // row n of W = col n of W^T
      acc[t] = WMMA_BF16(a, bf, acc[t]);
    }
  }

  // C/D layout: VGPR r -> M = r + 8*(lane>=16), N = lane&15
  int n = lane & 15, hi = lane >> 4;
#pragma unroll
  for (int t = 0; t < 4; ++t) {
    int ng = nb + 16 * t + n;
    float bv = bias[ng];
#pragma unroll
    for (int r = 0; r < 8; ++r) {
      int m = mr + r + 8 * hi;
      Pb[(size_t)m * F + ng] = (__bf16)(acc[t][r] + bv);
    }
  }
}

// ---- kernel 3: fused scores tile + mask + per-chunk top-50 -----------------
// grid = (N/16, N/CHUNK), block = 256 (8 waves). Block computes a 16xCHUNK
// score tile; wave w computes FOUR 16x16 tiles (cols 64w..64w+63) to maximize
// WMMA per streamed B-byte. The 16-row A slab (32 KB) is async-copied into
// LDS once and shared by all waves; after the K loop the same arena holds the
// masked score tile, from which each wave extracts top-50 for two rows.
__global__ __launch_bounds__(256)
void k_gemm2_topk(const __bf16* __restrict__ Pb, const __bf16* __restrict__ Mb,
                  float* __restrict__ cs, int* __restrict__ ci,
                  int N, int F, int nchunk) {
  int lane = threadIdx.x & 31, wave = threadIdx.x >> 5;   // wave in 0..7
  int r0 = blockIdx.x * 16;
  int c0 = blockIdx.y * CHUNK;

  // mask is j < i ; largest i in block is r0+15 -> chunk fully masked if c0 >= r0+15
  if (c0 >= r0 + 15) {
    for (int e = threadIdx.x; e < 16 * TOPK; e += blockDim.x) {
      int row = e / TOPK, slot = e % TOPK;
      size_t o = ((size_t)(r0 + row) * nchunk + blockIdx.y) * TOPK + slot;
      cs[o] = -__builtin_inff();
      ci[o] = c0 + slot;
    }
    return;
  }

  // arena: 33024 B; A slab (32768 B) during the K loop, score tile afterwards.
  __shared__ __align__(16) unsigned char arena[16 * LDS_STRIDE * 4];
  __bf16* Alds = (__bf16*)arena;
  float (*S)[LDS_STRIDE] = (float (*)[LDS_STRIDE])arena;

  async_slab_to_lds(Pb + (size_t)r0 * F, Alds, (16 * F) / 8,
                    (int)threadIdx.x, 256);

  f32x8 acc[4] = {};
  int cb = c0 + 64 * wave;
  for (int k0 = 0; k0 < F; k0 += 32) {
    if (k0 + 32 < F)  // pull next B K-slab toward L0 (global_prefetch_b8)
      __builtin_prefetch(Mb + (size_t)(cb + (lane & 15)) * F + k0 + 32, 0, 1);
    bf16x16 a = load_a_frag(Alds, F, 0, k0, lane);       // ds_load_b128 x2
#pragma unroll
    for (int t = 0; t < 4; ++t) {
      bf16x16 bf = load_b_frag(Mb, F, cb + 16 * t, k0, lane);
      acc[t] = WMMA_BF16(a, bf, acc[t]);
    }
  }
  __syncthreads();   // all A-slab reads complete before S overwrites the arena

  // masked epilogue -> LDS
  int n = lane & 15, hi = lane >> 4;
#pragma unroll
  for (int t = 0; t < 4; ++t) {
    int col = 64 * wave + 16 * t + n;
    int j = c0 + col;
#pragma unroll
    for (int r = 0; r < 8; ++r) {
      int m = r + 8 * hi;
      S[m][col] = (j < r0 + m) ? acc[t][r] : -__builtin_inff();
    }
  }
  __syncthreads();

  // wave w selects top-50 for rows 2w and 2w+1 (register-resident candidates)
  for (int rr = 0; rr < 2; ++rr) {
    int row = 2 * wave + rr;
    float vals[16];
#pragma unroll
    for (int s = 0; s < 16; ++s) vals[s] = S[row][lane + 32 * s];

    size_t obase = ((size_t)(r0 + row) * nchunk + blockIdx.y) * TOPK;
    for (int it = 0; it < TOPK; ++it) {
      float bv = vals[0]; int bs = 0;
#pragma unroll
      for (int s = 1; s < 16; ++s)
        if (vals[s] > bv) { bv = vals[s]; bs = s; }
      int bcol = lane + 32 * bs;

      float rv = bv; int rc = bcol;
#pragma unroll
      for (int off = 16; off > 0; off >>= 1) {
        float ov = __shfl_xor(rv, off, 32);
        int   oc = __shfl_xor(rc, off, 32);
        if (ov > rv || (ov == rv && oc < rc)) { rv = ov; rc = oc; }
      }
      if (lane == 0) { cs[obase + it] = rv; ci[obase + it] = c0 + rc; }
      if ((rc & 31) == lane) {      // winner lane retires its element
        int s = rc >> 5;
#pragma unroll
        for (int ss = 0; ss < 16; ++ss)
          if (ss == s) vals[ss] = -__builtin_inff();
      }
    }
  }
}

// ---- kernel 4: merge 16 chunk candidate lists -> final top-50 per row ------
// one wave per row; nchunk*TOPK (<= 800) candidates staged in LDS.
__global__ __launch_bounds__(32)
void k_merge(const float* __restrict__ cs, const int* __restrict__ ci,
             float* __restrict__ outS, int* __restrict__ outI, int nchunk) {
  int row = blockIdx.x, lane = threadIdx.x;
  int M = nchunk * TOPK;
  if (M > 16 * TOPK) M = 16 * TOPK;

  __shared__ float sv[16 * TOPK];
  __shared__ int   si[16 * TOPK];
  for (int e = lane; e < M; e += 32) {
    sv[e] = cs[(size_t)row * (nchunk * TOPK) + e];
    si[e] = ci[(size_t)row * (nchunk * TOPK) + e];
  }
  __syncthreads();

  for (int it = 0; it < TOPK; ++it) {
    float bv = -__builtin_inff(); int bg = 0x7fffffff; int be = 0;
    for (int e = lane; e < M; e += 32) {
      float v = sv[e]; int g = si[e];
      if (v > bv || (v == bv && g < bg)) { bv = v; bg = g; be = e; }
    }
    float rv = bv; int rg = bg; int re = be;
    for (int off = 16; off > 0; off >>= 1) {
      float ov = __shfl_xor(rv, off, 32);
      int   og = __shfl_xor(rg, off, 32);
      int   oe = __shfl_xor(re, off, 32);
      if (ov > rv || (ov == rv && og < rg)) { rv = ov; rg = og; re = oe; }
    }
    if (lane == 0) {
      outS[(size_t)row * TOPK + it] = rv;
      outI[(size_t)row * TOPK + it] = rg;
      sv[re] = -__builtin_inff();
    }
    __syncthreads();
  }
}

// ---------------------------------------------------------------------------
extern "C" void kernel_launch(void* const* d_in, const int* in_sizes, int n_in,
                              void* d_out, int out_size, void* d_ws, size_t ws_size,
                              hipStream_t stream) {
  const float* mentions = (const float*)d_in[0];
  const float* W        = (const float*)d_in[1];
  const float* bias     = (const float*)d_in[2];

  int F = in_sizes[2];                       // 1024
  long nM = in_sizes[0];                     // N*F
  int N = (int)(nM / F);                     // 8192
  int nchunk = N / CHUNK;                    // 16

  // workspace layout
  __bf16* Mb = (__bf16*)d_ws;                          // N*F bf16 (mentions)
  __bf16* Wb = Mb + (size_t)N * F;                     // F*F bf16
  __bf16* Pb = Wb + (size_t)F * F;                     // N*F bf16 (proj)
  float*  cs = (float*)(Pb + (size_t)N * F);           // N*nchunk*TOPK candidate scores
  int*    ci = (int*)(cs + (size_t)N * nchunk * TOPK); // candidate indices

  float* outS = (float*)d_out;
  int*   outI = (int*)(outS + (size_t)N * TOPK);

  long total = (long)N * F;
  k_convert<<<(unsigned)((total + 255) / 256), 256, 0, stream>>>(
      mentions, W, Mb, Wb, total, (long)F * F);

  dim3 g2(N / 16, F / 256);
  k_gemm1<<<g2, 128, 0, stream>>>(Mb, Wb, bias, Pb, N, F);

  dim3 g3(N / 16, nchunk);
  k_gemm2_topk<<<g3, 256, 0, stream>>>(Pb, Mb, cs, ci, N, F, nchunk);

  k_merge<<<N, 32, 0, stream>>>(cs, ci, outS, outI, nchunk);
}
